// Mambaspa_4020089389061
// MI455X (gfx1250) — compile-verified, matrix operand order checked by reading, not run on hardware
//
#include <hip/hip_runtime.h>
#include <hip/hip_bf16.h>
#include <stdint.h>

typedef __attribute__((ext_vector_type(16))) _Float16 v16h;
typedef __attribute__((ext_vector_type(8)))  _Float16 v8h;
typedef __attribute__((ext_vector_type(8)))  float    v8f;
typedef __attribute__((ext_vector_type(4)))  unsigned int v4u;
typedef __attribute__((ext_vector_type(4)))  int      v4i;
typedef __attribute__((ext_vector_type(8)))  int      v8i;

#define B_  8
#define C_  192
#define D_  192
#define H_  64
#define W_  64
#define L_  4096            // H_*W_
#define N_  16
#define K_  2
#define R_  12
#define CP_ 48              // R_+2N_=44 padded to 48 (3 WMMA row tiles)

// ---------------------------------------------------------------------------
// Tensor Data Mover: 2D tile load (16-bit elements) into LDS.
// D# per CDNA5 ISA ch.8: group0 = {count, lds_addr, global_addr, type=2};
// group1 = {data_size=1(2B), tensor dims, tile dims, dim0 stride}.
// Groups 2/3 zero (<=2D tensor). Tracked by TENSORcnt.
// ---------------------------------------------------------------------------
__device__ __forceinline__ void tdm_load_2d_f16(uint32_t lds_off, const void* gaddr,
                                                uint32_t tensor_d0, uint32_t tensor_d1,
                                                uint32_t tile_d0, uint32_t tile_d1,
                                                uint64_t stride_d0_elems) {
  uint64_t ga = (uint64_t)(uintptr_t)gaddr;
  v4u g0;
  g0[0] = 1u;                                            // count=1 (valid descriptor)
  g0[1] = lds_off;                                       // LDS byte address
  g0[2] = (uint32_t)(ga & 0xFFFFFFFFu);                  // global_addr[31:0]
  g0[3] = (uint32_t)((ga >> 32) & 0x1FFFFFFu) | (2u << 30);  // ga[56:32] | type=2
  v8i g1;
  g1[0] = (int)(1u << 16);                               // data_size=1 -> 2 bytes
  g1[1] = (int)((tensor_d0 & 0xFFFFu) << 16);            // tensor_dim0[15:0]
  g1[2] = (int)(((tensor_d0 >> 16) & 0xFFFFu) |
                ((tensor_d1 & 0xFFFFu) << 16));          // dim0[31:16] | dim1[15:0]
  g1[3] = (int)(((tensor_d1 >> 16) & 0xFFFFu) |
                ((tile_d0 & 0xFFFFu) << 16));            // dim1[31:16] | tile_dim0
  g1[4] = (int)(tile_d1 & 0xFFFFu);                      // tile_dim1 (tile_dim2=0)
  g1[5] = (int)(uint32_t)(stride_d0_elems & 0xFFFFFFFFu);// tensor_dim0_stride[31:0]
  g1[6] = (int)((stride_d0_elems >> 32) & 0xFFFFu);      // stride[47:32]
  g1[7] = 0;
  v4i z4 = {0, 0, 0, 0};
#if __clang_major__ >= 23
  v8i z8 = {0, 0, 0, 0, 0, 0, 0, 0};
  __builtin_amdgcn_tensor_load_to_lds(g0, g1, z4, z4, z8, 0);
#else
  __builtin_amdgcn_tensor_load_to_lds(g0, g1, z4, z4, 0);
#endif
}

// ---------------------------------------------------------------------------
// WMMA fragment helpers (CDNA5 16-bit 16x32 layout, ISA 7.12.2).
// Fragment elements [0..7]  = K = 8*half .. 8*half+7   (contiguous)
//                   [8..15] = K = 16+8*half .. +7      (contiguous)
// -> two 128-bit loads per lane from any K-contiguous row.
// ---------------------------------------------------------------------------
__device__ __forceinline__ v16h frag_from_row(const _Float16* row, int half) {
  v8h lo = *(const v8h*)(row + 8 * half);
  v8h hi = *(const v8h*)(row + 16 + 8 * half);
  return __builtin_shufflevector(lo, hi, 0, 1, 2, 3, 4, 5, 6, 7, 8, 9, 10, 11, 12, 13,
                                 14, 15);
}

// ---------------------------------------------------------------------------
// Kernel 1: precision prep.  xh_t is x TRANSPOSED to (B, L, C) pixel-major so
// TDM panels land d-contiguous in LDS.
// ---------------------------------------------------------------------------
__global__ void mamba_prep_kernel(const float* __restrict__ x,
                                  const float* __restrict__ wi,
                                  const float* __restrict__ wo,
                                  const float* __restrict__ xpw,
                                  _Float16* __restrict__ xh_t,
                                  _Float16* __restrict__ wih,
                                  _Float16* __restrict__ woh,
                                  _Float16* __restrict__ xpwh) {
  long i = (long)blockIdx.x * blockDim.x + threadIdx.x;
  if (i < (long)B_ * C_ * L_) {
    int p = (int)(i % L_);
    int c = (int)((i / L_) % C_);
    int b = (int)(i / ((long)C_ * L_));
    xh_t[((long)b * L_ + p) * C_ + c] = (_Float16)x[i];  // coalesced read
  }
  if (i < D_ * C_) {
    wih[i] = (_Float16)wi[i];
    woh[i] = (_Float16)wo[i];
  }
  if (i < (long)K_ * CP_ * D_) {
    int col = (int)(i % D_);
    int c   = (int)((i / D_) % CP_);
    int k   = (int)(i / (D_ * CP_));
    xpwh[i] = (c < (R_ + 2 * N_)) ? (_Float16)xpw[((long)k * (R_ + 2 * N_) + c) * D_ + col]
                                  : (_Float16)0.0f;
  }
}

// ---------------------------------------------------------------------------
// Shared 192x192 @ (192 x L) GEMM:  dst[b, m, p] = sum_k Aw[m,k] * Bsrc[b,p,k]
// Block = 384 threads = 12 waves; one TDM panel (16 pixels x 192) feeds all
// 12 output row-tiles.  Used for in_proj (Bsrc=xh_t) and out_proj (Bsrc=yln).
// ---------------------------------------------------------------------------
__global__ void mamba_gemm192_kernel(const _Float16* __restrict__ Bsrc,
                                     const _Float16* __restrict__ Aw,
                                     float* __restrict__ dst) {
  __shared__ _Float16 shB[16 * 192];  // [n][k], 6 KB
  int pt = blockIdx.x & 255;
  int b  = blockIdx.x >> 8;
  int p0 = pt * 16;
  int wave = threadIdx.x >> 5;        // 0..11 -> output row tile
  int lane = threadIdx.x & 31;
  int d0 = wave * 16;

  if (threadIdx.x < 32) {             // wave 0 issues the panel DMA
    uint32_t lds_off = (uint32_t)(uintptr_t)&shB[0];
    tdm_load_2d_f16(lds_off, Bsrc + ((long)b * L_ + p0) * 192, 192, 16, 192, 16, 192);
    __builtin_amdgcn_s_wait_tensorcnt(0);
  }
  __syncthreads();

  int n = lane & 15, half = lane >> 4;
  const _Float16* arow = Aw + (long)d0 * 192 + (long)(lane & 15) * 192;
  const _Float16* brow = &shB[n * 192];
  v8f acc = {};
#pragma unroll
  for (int kk = 0; kk < 192 / 32; ++kk) {
    int c0 = kk * 32;
    if (kk + 1 < 192 / 32)
      __builtin_prefetch(arow + c0 + 32, 0, 1);   // global_prefetch_b8
    v16h a  = frag_from_row(arow + c0, half);
    v16h bm = frag_from_row(brow + c0, half);
    acc = __builtin_amdgcn_wmma_f32_16x16x32_f16(false, a, false, bm, (short)0, acc,
                                                 false, false);
  }
  float* dp = dst + (long)b * 192 * L_ + (long)d0 * L_ + p0;
#pragma unroll
  for (int v = 0; v < 8; ++v) dp[(long)(v + 8 * half) * L_ + n] = acc[v];
}

// ---------------------------------------------------------------------------
// Kernel 3: depthwise 3x3 conv + bias + SiLU -> xch_t (B, L, D) f16
// ---------------------------------------------------------------------------
__global__ void mamba_dwconv_silu_kernel(const float* __restrict__ xin,
                                         const float* __restrict__ cw,
                                         const float* __restrict__ cb,
                                         _Float16* __restrict__ xch_t) {
  long i = (long)blockIdx.x * blockDim.x + threadIdx.x;
  if (i >= (long)B_ * D_ * L_) return;
  int p = (int)(i % L_);
  int d = (int)((i / L_) % D_);
  int b = (int)(i / ((long)D_ * L_));
  int h = p >> 6, w = p & 63;
  const float* src = xin + ((long)b * D_ + d) * L_;
  const float* wz  = cw + d * 9;
  float s = cb[d];
#pragma unroll
  for (int dy = -1; dy <= 1; ++dy) {
    int hh = h + dy;
    if (hh < 0 || hh >= H_) continue;
#pragma unroll
    for (int dx = -1; dx <= 1; ++dx) {
      int ww = w + dx;
      if (ww < 0 || ww >= W_) continue;
      s += wz[(dy + 1) * 3 + (dx + 1)] * src[hh * W_ + ww];
    }
  }
  float sig = 1.0f / (1.0f + __expf(-s));
  xch_t[((long)b * L_ + p) * D_ + d] = (_Float16)(s * sig);
}

// ---------------------------------------------------------------------------
// Kernel 4: x_dbl[b,k,c,l] = x_proj_w[k] (48x192 padded) @ xs[b,k] (192 x L).
// Pixel-major xch_t means even the reversed direction is a contiguous 16-row
// memory block: TDM loads it, reversal = reading LDS row (15-n).
// Block = 96 threads = 3 waves (one per 16-row c-tile).
// ---------------------------------------------------------------------------
__global__ void mamba_gemm_xdbl_kernel(const _Float16* __restrict__ xch_t,
                                       const _Float16* __restrict__ xpwh,
                                       float* __restrict__ xdbl) {
  __shared__ _Float16 shB[16 * 192];  // [row][d], 6 KB
  int pt = blockIdx.x & 255;
  int k  = (blockIdx.x >> 8) & 1;
  int b  = blockIdx.x >> 9;
  int p0 = pt * 16;
  int wave = threadIdx.x >> 5;        // 0..2 -> c-tile
  int lane = threadIdx.x & 31;
  int c0 = wave * 16;

  int q0 = (k == 0) ? p0 : (L_ - 16 - p0);  // memory row base of the window
  if (threadIdx.x < 32) {
    uint32_t lds_off = (uint32_t)(uintptr_t)&shB[0];
    tdm_load_2d_f16(lds_off, xch_t + ((long)b * L_ + q0) * D_, D_, 16, D_, 16, D_);
    __builtin_amdgcn_s_wait_tensorcnt(0);
  }
  __syncthreads();

  int n = lane & 15, half = lane >> 4;
  int row = (k == 0) ? n : (15 - n);        // reversal via LDS row flip
  const _Float16* arow = xpwh + ((long)k * CP_ + c0 + n) * D_;  // A row m=n per lane
  const _Float16* brow = &shB[row * D_];
  v8f acc = {};
#pragma unroll
  for (int kk = 0; kk < D_ / 32; ++kk) {
    int d0 = kk * 32;
    v16h a  = frag_from_row(arow + d0, half);
    v16h bm = frag_from_row(brow + d0, half);
    acc = __builtin_amdgcn_wmma_f32_16x16x32_f16(false, a, false, bm, (short)0, acc,
                                                 false, false);
  }
  float* dp = xdbl + ((long)(b * K_ + k) * CP_ + c0) * L_ + p0;
#pragma unroll
  for (int v = 0; v < 8; ++v) dp[(long)(v + 8 * half) * L_ + n] = acc[v];
}

// ---------------------------------------------------------------------------
// Kernel 5: fused dt-projection + softplus + selective scan.
// One thread per (b,k,d); u reads and ys writes are d-contiguous (coalesced).
// ---------------------------------------------------------------------------
__global__ void mamba_scan_kernel(const _Float16* __restrict__ xch_t,
                                  const float* __restrict__ xdbl,
                                  const float* __restrict__ dtw,
                                  const float* __restrict__ dtb,
                                  const float* __restrict__ alogs,
                                  const float* __restrict__ dvec,
                                  float* __restrict__ ys) {
  int t = blockIdx.x * blockDim.x + threadIdx.x;
  if (t >= B_ * K_ * D_) return;
  int d = t % D_;
  int k = (t / D_) & 1;
  int b = t / (K_ * D_);
  int kd = k * D_ + d;

  float w[R_];
#pragma unroll
  for (int r = 0; r < R_; ++r) w[r] = dtw[kd * R_ + r];
  float bias = dtb[kd];
  float An[N_], h[N_];
#pragma unroll
  for (int n = 0; n < N_; ++n) {
    An[n] = -__expf(alogs[kd * N_ + n]);
    h[n] = 0.0f;
  }
  float Dv = dvec[kd];

  const _Float16* u_base = xch_t + (long)b * L_ * D_ + d;
  const float* db = xdbl + (long)(b * K_ + k) * CP_ * L_;
  float* yo = ys + ((long)(b * K_ + k) * L_) * D_ + d;  // (b,k,l,d)

  for (int l = 0; l < L_; ++l) {
    int pos = k ? (L_ - 1 - l) : l;
    float u = (float)u_base[(long)pos * D_];
    float acc = bias;
#pragma unroll
    for (int r = 0; r < R_; ++r) acc += w[r] * db[(long)r * L_ + l];
    float dt = (acc > 20.0f) ? acc : __logf(1.0f + __expf(acc));  // softplus
    float du = dt * u;
    float y = 0.0f;
#pragma unroll
    for (int n = 0; n < N_; ++n) {
      float Bv = db[(long)(R_ + n) * L_ + l];
      float Cv = db[(long)(R_ + N_ + n) * L_ + l];
      h[n] = __expf(dt * An[n]) * h[n] + du * Bv;
      y += h[n] * Cv;
    }
    yo[(long)l * D_] = y + Dv * u;
  }
}

// ---------------------------------------------------------------------------
// Kernel 6: combine directions + LayerNorm over D (gate == 1 exactly:
// softmax over a singleton axis).  Output f16 (b,p,d) for out-GEMM.
// ---------------------------------------------------------------------------
__global__ void mamba_combine_ln_kernel(const float* __restrict__ ys,
                                        const float* __restrict__ g,
                                        const float* __restrict__ be,
                                        _Float16* __restrict__ yln) {
  int bp = blockIdx.x;
  int d  = threadIdx.x;
  int b  = bp / L_;
  int p  = bp % L_;
  float v = ys[((long)(b * K_ + 0) * L_ + p) * D_ + d] +
            ys[((long)(b * K_ + 1) * L_ + (L_ - 1 - p)) * D_ + d];
  __shared__ float s1[D_], s2[D_];
  __shared__ float mu_s, rs_s;
  s1[d] = v;
  s2[d] = v * v;
  __syncthreads();
  for (int off = 96; off > 2; off >>= 1) {  // 96,48,24,12,6,3
    if (d < off) {
      s1[d] += s1[d + off];
      s2[d] += s2[d + off];
    }
    __syncthreads();
  }
  if (d == 0) {
    float sum = s1[0] + s1[1] + s1[2];
    float sq  = s2[0] + s2[1] + s2[2];
    float mu  = sum / (float)D_;
    float var = sq / (float)D_ - mu * mu;
    mu_s = mu;
    rs_s = rsqrtf(var + 1e-5f);
  }
  __syncthreads();
  float yn = (v - mu_s) * rs_s * g[d] + be[d];
  yln[(long)bp * D_ + d] = (_Float16)yn;
}

// ---------------------------------------------------------------------------
// Host launcher
// ---------------------------------------------------------------------------
extern "C" void kernel_launch(void* const* d_in, const int* in_sizes, int n_in,
                              void* d_out, int out_size, void* d_ws, size_t ws_size,
                              hipStream_t stream) {
  (void)in_sizes; (void)n_in; (void)out_size; (void)ws_size;
  const float* x    = (const float*)d_in[0];
  const float* wi   = (const float*)d_in[1];   // in_proj_w (D,C)
  const float* wo   = (const float*)d_in[2];   // out_proj_w (C,D)
  // d_in[3] skip_w, d_in[4] skip_b: gate = softmax over singleton axis == 1 -> unused
  const float* cw   = (const float*)d_in[5];   // conv_w (D,1,3,3)
  const float* cb   = (const float*)d_in[6];   // conv_b (D)
  const float* xpw  = (const float*)d_in[7];   // x_proj_w (K,44,D)
  const float* dtw  = (const float*)d_in[8];   // dt_proj_w (K,D,R)
  const float* dtb  = (const float*)d_in[9];   // dt_proj_b (K,D)
  const float* alog = (const float*)d_in[10];  // A_logs (K*D,N)
  const float* dvec = (const float*)d_in[11];  // Ds (K*D)
  const float* lng  = (const float*)d_in[12];
  const float* lnb  = (const float*)d_in[13];
  float* out = (float*)d_out;

  char* ws = (char*)d_ws;
  size_t off = 0;
  auto alloc = [&](size_t bytes) {
    char* p = ws + off;
    off = (off + bytes + 255) & ~(size_t)255;
    return p;
  };
  const long BCL = (long)B_ * C_ * L_;               // 6.29M elements
  _Float16* xh_t  = (_Float16*)alloc(BCL * 2);       // x transposed (B,L,C) f16
  _Float16* wih   = (_Float16*)alloc((size_t)D_ * C_ * 2);
  _Float16* woh   = (_Float16*)alloc((size_t)C_ * D_ * 2);
  _Float16* xpwh  = (_Float16*)alloc((size_t)K_ * CP_ * D_ * 2);
  float*    xin   = (float*)alloc(BCL * 4);          // in_proj output (B,D,L)
  _Float16* xch_t = (_Float16*)alloc(BCL * 2);       // conv+silu output (B,L,D) f16
  float*    xdbl  = (float*)alloc((size_t)B_ * K_ * CP_ * L_ * 4);
  float*    ysb   = (float*)alloc((size_t)B_ * K_ * L_ * D_ * 4);  // (b,k,l,d)
  _Float16* yln   = (_Float16*)alloc(BCL * 2);       // LN output (b,p,d) f16

  const int eltBlocks = (int)((BCL + 255) / 256);

  mamba_prep_kernel<<<eltBlocks, 256, 0, stream>>>(x, wi, wo, xpw, xh_t, wih, woh, xpwh);

  // in_proj: xin = wih @ x[b]
  mamba_gemm192_kernel<<<B_ * (L_ / 16), 384, 0, stream>>>(xh_t, wih, xin);

  mamba_dwconv_silu_kernel<<<eltBlocks, 256, 0, stream>>>(xin, cw, cb, xch_t);

  mamba_gemm_xdbl_kernel<<<B_ * K_ * (L_ / 16), 96, 0, stream>>>(xch_t, xpwh, xdbl);

  mamba_scan_kernel<<<(B_ * K_ * D_ + 255) / 256, 256, 0, stream>>>(xch_t, xdbl, dtw,
                                                                    dtb, alog, dvec,
                                                                    ysb);

  mamba_combine_ln_kernel<<<B_ * L_, D_, 0, stream>>>(ysb, lng, lnb, yln);

  // out_proj: out = woh @ y[b]
  mamba_gemm192_kernel<<<B_ * (L_ / 16), 384, 0, stream>>>(yln, woh, out);
}
